// CrossAttention_85856396247624
// MI455X (gfx1250) — matrix operand order repaired and checked by reading, not compiled, not run on hardware
//
#include <hip/hip_runtime.h>

// ---------------------------------------------------------------------------
// CrossAttention pipeline for MI455X (gfx1250, wave32, WMMA).
// All heavy math runs through one bf16 WMMA GEMM kernel
// (v_wmma_f32_16x16x32_bf16) with double-buffered LDS tiles so global->LDS
// staging of tile t+1 overlaps the 8 WMMAs of tile t. Attention scores are
// materialized in bf16 (64MB, fits the 192MB L2), softmax in place,
// P@V -> f32 d_out. Workspace assumption: ~332MB of d_ws.
// ---------------------------------------------------------------------------

typedef __attribute__((ext_vector_type(16))) __bf16 v16bf;
typedef __attribute__((ext_vector_type(8)))  float  v8f;

typedef unsigned short u16;
typedef unsigned int   u32;

__device__ __forceinline__ u16 f2bf(float f) {
  u32 u = __float_as_uint(f);
  u32 r = (u + 0x7FFFu + ((u >> 16) & 1u)) >> 16;   // round-to-nearest-even
  return (u16)r;
}
__device__ __forceinline__ float bf2f(u16 h) {
  return __uint_as_float(((u32)h) << 16);
}

#define TM 128
#define TN 128
#define TK 32
#define LSTR 40   // LDS row stride in bf16 elems (80B: 16B-aligned, 16 distinct banks for m=0..15)

// C[M,N] = act(scale * A[M,K] @ Bt[N,K]^T + bias), bf16 A/Bt, f32 accum.
// Batched via blockIdx.z with element strides sA/sB/sC. Outputs bf16 (Cb)
// and/or f32 (Cf).
__global__ __launch_bounds__(256)
void gemm_bf16_wmma(const u16* __restrict__ A, long long sA,
                    const u16* __restrict__ Bt, long long sB,
                    const float* __restrict__ bias, float scale, int relu,
                    u16* __restrict__ Cb, float* __restrict__ Cf, long long sC,
                    int M, int N, int K)
{
  __shared__ __align__(16) u16 As[2][TM * LSTR];
  __shared__ __align__(16) u16 Bs[2][TN * LSTR];

  const int z = blockIdx.z;
  A  += (long long)z * sA;
  Bt += (long long)z * sB;

  const int tid   = threadIdx.x;
  const int lane  = tid & 31;
  const int wave  = tid >> 5;      // 0..7
  const int waveM = wave >> 1;     // 0..3 : 32 rows each
  const int waveN = wave & 1;      // 0..1 : 64 cols each
  const int blockM = blockIdx.y * TM;
  const int blockN = blockIdx.x * TN;

  const int m0 = lane & 15;
  const int hi = lane >> 4;

  // K offsets per fragment VGPR pair, per ISA 16-bit layouts:
  // A 16x32: lanes 0-15 hold K{0..7,16..23}, lanes 16-31 hold K{8..15,24..31}
  // B 32x16: same K order as A (lanes 0-15 K{0..7,16..23}, lanes 16-31 K{8..15,24..31})
  int kbA[8], kbB[8];
#pragma unroll
  for (int i = 0; i < 8; ++i) {
    kbA[i] = (i < 4) ? (2 * i + 8 * hi) : (16 + 2 * (i - 4) + 8 * hi);
    kbB[i] = (i < 4) ? (2 * i + 8 * hi) : (16 + 2 * (i - 4) + 8 * hi);
  }

  v8f acc[2][4];
#pragma unroll
  for (int im = 0; im < 2; ++im)
#pragma unroll
    for (int jn = 0; jn < 4; ++jn)
#pragma unroll
      for (int r = 0; r < 8; ++r) acc[im][jn][r] = 0.0f;

  // Cooperative tile stage: 128 rows x 32 bf16 (= 4 uint4 per row), A and Bt.
  auto stage = [&](int kc, int buf) {
#pragma unroll
    for (int r = 0; r < 2; ++r) {
      int idx = tid + r * 256;
      int row = idx >> 2;
      int seg = idx & 3;
      const uint4* srcA = (const uint4*)(A + (long long)(blockM + row) * K + kc) + seg;
      const uint4* srcB = (const uint4*)(Bt + (long long)(blockN + row) * K + kc) + seg;
      *(uint4*)&As[buf][row * LSTR + seg * 8] = *srcA;
      *(uint4*)&Bs[buf][row * LSTR + seg * 8] = *srcB;
      // Speculative prefetch of the tile after next (global_prefetch_b8);
      // unguarded on purpose: speculative prefetch past the end is dropped.
      __builtin_prefetch((const void*)(A + (long long)(blockM + row) * K + kc + TK), 0, 3);
      __builtin_prefetch((const void*)(Bt + (long long)(blockN + row) * K + kc + TK), 0, 3);
    }
  };

  const int T = K / TK;
  stage(0, 0);

  for (int t = 0; t < T; ++t) {
    __syncthreads();  // tile t stores visible; protects WAR on buffer reuse
    if (t + 1 < T) stage((t + 1) * TK, (t + 1) & 1);

    const u16* Ab = As[t & 1];
    const u16* Bb = Bs[t & 1];

    union { v16bf v; u16 u[16]; } af[2], bfrg[4];
#pragma unroll
    for (int im = 0; im < 2; ++im) {
      const u16* src = &Ab[(waveM * 32 + im * 16 + m0) * LSTR];
#pragma unroll
      for (int i = 0; i < 8; ++i) {
        af[im].u[2 * i]     = src[kbA[i]];
        af[im].u[2 * i + 1] = src[kbA[i] + 1];
      }
    }
#pragma unroll
    for (int jn = 0; jn < 4; ++jn) {
      const u16* src = &Bb[(waveN * 64 + jn * 16 + m0) * LSTR];
#pragma unroll
      for (int i = 0; i < 8; ++i) {
        bfrg[jn].u[2 * i]     = src[kbB[i]];
        bfrg[jn].u[2 * i + 1] = src[kbB[i] + 1];
      }
    }
#pragma unroll
    for (int im = 0; im < 2; ++im)
#pragma unroll
      for (int jn = 0; jn < 4; ++jn)
        acc[im][jn] = __builtin_amdgcn_wmma_f32_16x16x32_bf16(
            false, af[im].v, false, bfrg[jn].v, (short)0, acc[im][jn], false, false);
  }

  float* Cfp = Cf ? Cf + (long long)z * sC : nullptr;
  u16*   Cbp = Cb ? Cb + (long long)z * sC : nullptr;
#pragma unroll
  for (int im = 0; im < 2; ++im) {
#pragma unroll
    for (int jn = 0; jn < 4; ++jn) {
      int col = blockN + waveN * 64 + jn * 16 + m0;
      float bv = bias ? bias[col] : 0.0f;
#pragma unroll
      for (int r = 0; r < 8; ++r) {   // D layout: row = r + 8*hi, col = lane&15
        int row = blockM + waveM * 32 + im * 16 + 8 * hi + r;
        float v = acc[im][jn][r] * scale + bv;
        if (relu) v = fmaxf(v, 0.0f);
        long long off = (long long)row * N + col;
        if (Cfp) Cfp[off] = v;
        if (Cbp) Cbp[off] = f2bf(v);
      }
    }
  }
}

// ---------------------------------------------------------------------------
__global__ __launch_bounds__(256)
void cast_f32_bf16(const float* __restrict__ in, u16* __restrict__ out, long long n) {
  long long i = (long long)blockIdx.x * 1024 + threadIdx.x;
#pragma unroll
  for (int r = 0; r < 4; ++r) {
    long long j = i + r * 256;
    if (j < n) out[j] = f2bf(in[j]);
  }
}

// out[c*R + r] = bf16(in[r*C + c])   (weights: R=K, C=N -> Wt is N x K)
__global__ __launch_bounds__(256)
void transpose_cast_f32_bf16(const float* __restrict__ in, u16* __restrict__ out,
                             int R, int C) {
  __shared__ float tile[32][33];
  int r0 = blockIdx.y * 32, c0 = blockIdx.x * 32;
  int tx = threadIdx.x & 31, ty = threadIdx.x >> 5;
  for (int i = ty; i < 32; i += 8)
    tile[i][tx] = in[(long long)(r0 + i) * C + (c0 + tx)];
  __syncthreads();
  for (int i = ty; i < 32; i += 8)
    out[(long long)(c0 + i) * R + (r0 + tx)] = f2bf(tile[tx][i]);
}

// batched bf16 transpose: out[z][c*R + r] = in[z][r*C + c]
__global__ __launch_bounds__(256)
void transpose_bf16(const u16* __restrict__ in, u16* __restrict__ out, int R, int C) {
  __shared__ u16 tile[32][33];
  long long base = (long long)blockIdx.z * R * C;
  int r0 = blockIdx.y * 32, c0 = blockIdx.x * 32;
  int tx = threadIdx.x & 31, ty = threadIdx.x >> 5;
  for (int i = ty; i < 32; i += 8)
    tile[i][tx] = in[base + (long long)(r0 + i) * C + (c0 + tx)];
  __syncthreads();
  for (int i = ty; i < 32; i += 8)
    out[base + (long long)(c0 + i) * R + (r0 + tx)] = tile[tx][i];
}

// In-place row softmax over bf16 scores. One block per row, 256 threads,
// wave32 shuffle reductions + cross-wave LDS combine.
__global__ __launch_bounds__(256)
void softmax_bf16_inplace(u16* __restrict__ S, int cols) {
  long long row = blockIdx.x;
  u16* s = S + row * (long long)cols;
  __shared__ float red[8];
  const int tid = threadIdx.x, lane = tid & 31, wv = tid >> 5;

  float m = -3.4e38f;
  for (int c = tid; c < cols; c += 256) m = fmaxf(m, bf2f(s[c]));
#pragma unroll
  for (int o = 16; o > 0; o >>= 1) m = fmaxf(m, __shfl_xor(m, o, 32));
  if (lane == 0) red[wv] = m;
  __syncthreads();
  float rm = red[0];
#pragma unroll
  for (int i = 1; i < 8; ++i) rm = fmaxf(rm, red[i]);
  __syncthreads();

  float sum = 0.0f;
  for (int c = tid; c < cols; c += 256) {
    float e = __expf(bf2f(s[c]) - rm);
    s[c] = f2bf(e);
    sum += e;
  }
#pragma unroll
  for (int o = 16; o > 0; o >>= 1) sum += __shfl_xor(sum, o, 32);
  if (lane == 0) red[wv] = sum;
  __syncthreads();
  float ts = 0.0f;
#pragma unroll
  for (int i = 0; i < 8; ++i) ts += red[i];
  float inv = 1.0f / ts;
  for (int c = tid; c < cols; c += 256) s[c] = f2bf(bf2f(s[c]) * inv);
}

// ---------------------------------------------------------------------------
static inline void launch_gemm(hipStream_t stream,
                               const u16* A, long long sA,
                               const u16* Bt, long long sB,
                               const float* bias, float scale, int relu,
                               u16* Cb, float* Cf, long long sC,
                               int M, int N, int K, int batch) {
  dim3 grid(N / TN, M / TM, batch);
  gemm_bf16_wmma<<<grid, dim3(256), 0, stream>>>(A, sA, Bt, sB, bias, scale,
                                                 relu, Cb, Cf, sC, M, N, K);
}

extern "C" void kernel_launch(void* const* d_in, const int* in_sizes, int n_in,
                              void* d_out, int out_size, void* d_ws, size_t ws_size,
                              hipStream_t stream) {
  (void)in_sizes; (void)n_in; (void)out_size; (void)ws_size;
  const int B = 8, Sq = 2048, Skv = 2048, D = 1024, DB = 512;
  const int Mx = B * Sq;   // 16384
  const int My = B * Skv;  // 16384

  const float* x  = (const float*)d_in[0];
  const float* y  = (const float*)d_in[1];
  const float* W1 = (const float*)d_in[2];  const float* b1 = (const float*)d_in[3];
  const float* W2 = (const float*)d_in[4];  const float* b2 = (const float*)d_in[5];
  const float* W3 = (const float*)d_in[6];  const float* b3 = (const float*)d_in[7];
  const float* W4 = (const float*)d_in[8];  const float* b4 = (const float*)d_in[9];
  const float* Wq = (const float*)d_in[10]; const float* bq = (const float*)d_in[11];
  const float* Wk = (const float*)d_in[12]; const float* bk = (const float*)d_in[13];
  const float* Wv = (const float*)d_in[14]; const float* bv = (const float*)d_in[15];

  char* w = (char*)d_ws;
  auto carve = [&](size_t bytes) -> char* {
    char* p = w;
    w += (bytes + 255) & ~(size_t)255;
    return p;
  };
  u16* W1t = (u16*)carve((size_t)D * D * 2);
  u16* W2t = (u16*)carve((size_t)D * DB * 2);
  u16* W3t = (u16*)carve((size_t)DB * D * 2);
  u16* W4t = (u16*)carve((size_t)D * D * 2);
  u16* Wqt = (u16*)carve((size_t)D * D * 2);
  u16* Wkt = (u16*)carve((size_t)D * D * 2);
  u16* Wvt = (u16*)carve((size_t)D * D * 2);
  u16* xb  = (u16*)carve((size_t)Mx * D * 2);
  u16* yb  = (u16*)carve((size_t)My * D * 2);
  u16* hA  = (u16*)carve((size_t)Mx * D * 2);
  u16* hB  = (u16*)carve((size_t)Mx * D * 2);
  u16* qb  = (u16*)carve((size_t)Mx * D * 2);
  u16* kb  = (u16*)carve((size_t)My * D * 2);
  u16* vb  = (u16*)carve((size_t)My * D * 2);
  u16* vt  = (u16*)carve((size_t)My * D * 2);
  u16* S   = (u16*)carve((size_t)B * Sq * Skv * 2);  // scores, then P in-place

  dim3 blk(256);
  // Weight transposes (Wt is N x K, so B-fragment K-pairs are contiguous).
  transpose_cast_f32_bf16<<<dim3(D / 32,  D / 32),  blk, 0, stream>>>(W1, W1t, D,  D);
  transpose_cast_f32_bf16<<<dim3(DB / 32, D / 32),  blk, 0, stream>>>(W2, W2t, D,  DB);
  transpose_cast_f32_bf16<<<dim3(D / 32,  DB / 32), blk, 0, stream>>>(W3, W3t, DB, D);
  transpose_cast_f32_bf16<<<dim3(D / 32,  D / 32),  blk, 0, stream>>>(W4, W4t, D,  D);
  transpose_cast_f32_bf16<<<dim3(D / 32,  D / 32),  blk, 0, stream>>>(Wq, Wqt, D,  D);
  transpose_cast_f32_bf16<<<dim3(D / 32,  D / 32),  blk, 0, stream>>>(Wk, Wkt, D,  D);
  transpose_cast_f32_bf16<<<dim3(D / 32,  D / 32),  blk, 0, stream>>>(Wv, Wvt, D,  D);

  // Activation casts.
  cast_f32_bf16<<<dim3((Mx * (long long)D) / 1024), blk, 0, stream>>>(x, xb, (long long)Mx * D);
  cast_f32_bf16<<<dim3((My * (long long)D) / 1024), blk, 0, stream>>>(y, yb, (long long)My * D);

  // 4-layer MLP, fused bias+ReLU.
  launch_gemm(stream, xb, 0, W1t, 0, b1, 1.0f, 1, hA, nullptr, 0, Mx, D,  D,  1);
  launch_gemm(stream, hA, 0, W2t, 0, b2, 1.0f, 1, hB, nullptr, 0, Mx, DB, D,  1);
  launch_gemm(stream, hB, 0, W3t, 0, b3, 1.0f, 1, hA, nullptr, 0, Mx, D,  DB, 1);
  launch_gemm(stream, hA, 0, W4t, 0, b4, 1.0f, 1, hB, nullptr, 0, Mx, D,  D,  1);

  // q/k/v projections (bias, no ReLU).
  launch_gemm(stream, hB, 0, Wqt, 0, bq, 1.0f, 0, qb, nullptr, 0, Mx, D, D, 1);
  launch_gemm(stream, yb, 0, Wkt, 0, bk, 1.0f, 0, kb, nullptr, 0, My, D, D, 1);
  launch_gemm(stream, yb, 0, Wvt, 0, bv, 1.0f, 0, vb, nullptr, 0, My, D, D, 1);

  // scores = (1/sqrt(D)) * q @ k^T   (k row-major [Skv][D] == Bt layout)
  const float scale = 0.03125f;  // 1/sqrt(1024)
  launch_gemm(stream, qb, (long long)Sq * D, kb, (long long)Skv * D,
              nullptr, scale, 0, S, nullptr, (long long)Sq * Skv,
              Sq, Skv, D, B);

  // softmax rows in place (S -> P).
  softmax_bf16_inplace<<<dim3(B * Sq), blk, 0, stream>>>(S, Skv);

  // v^T per batch: vt[d][s] = v[s][d]
  transpose_bf16<<<dim3(D / 32, Skv / 32, B), blk, 0, stream>>>(vb, vt, Skv, D);

  // out = P @ V  -> f32 d_out
  launch_gemm(stream, S, (long long)Sq * Skv, vt, (long long)D * Skv,
              nullptr, 1.0f, 0, nullptr, (float*)d_out, (long long)Sq * D,
              Sq, D, Skv, B);
}
